// VanillaLSTM_16552803959447
// MI455X (gfx1250) — matrix operand order repaired and checked by reading, not compiled, not run on hardware
//
#include <hip/hip_runtime.h>

// ---------------------------------------------------------------------------
// VanillaLSTM forward for MI455X (gfx1250, wave32, WMMA).
//
// B=4096, T=1024, D_IN=2, EMD=CELL=128, G=4*CELL=512, D_OUT=5.
// Grid: 256 WGs, each owns 16 batch rows and runs the full T loop.
// Block: 512 threads = 16 waves; wave w owns gate columns [w*32, w*32+32).
// Weights (W_ih, W_hh) live in VGPRs as WMMA B-fragments for the whole loop.
// h/c state lives in LDS. Gate GEMMs: v_wmma_f32_16x16x32_bf16.
// Output projection split 4-ways per dot product + __shfl_xor quad reduce.
// ---------------------------------------------------------------------------

typedef __attribute__((ext_vector_type(16))) __bf16 v16bf;
typedef __attribute__((ext_vector_type(8)))  __bf16 v8bf;
typedef __attribute__((ext_vector_type(8)))  float  v8f;

namespace {
constexpr int kB    = 4096;
constexpr int kT    = 1024;
constexpr int kDIN  = 2;
constexpr int kEMD  = 128;
constexpr int kCELL = 128;
constexpr int kG    = 512;   // 4*CELL
constexpr int kDOUT = 5;
constexpr int kROWS = 16;    // batch rows per WG (WMMA M)
constexpr int kTHREADS = 512; // 16 waves
}

__device__ __forceinline__ float fast_sigmoid(float x) {
  return 1.0f / (1.0f + __expf(-x));
}
__device__ __forceinline__ float fast_tanh(float x) {
  // tanh(x) = 1 - 2/(exp(2x)+1); saturates correctly at +-1.
  return 1.0f - 2.0f / (__expf(2.0f * x) + 1.0f);
}

// Load a 16x32 bf16 A-fragment from an LDS tile (row-major, `stride` halfs).
// ISA layout (16-bit A 16x32): lane L holds row M=L&15;
// lanes 0-15:  K in {kt*32+0..7,  kt*32+16..23}
// lanes 16-31: K in {kt*32+8..15, kt*32+24..31}
__device__ __forceinline__ v16bf load_a16(const __bf16* base, int stride,
                                          int kt, int lane) {
  const int m  = lane & 15;
  const int kb = kt * 32 + ((lane >> 4) << 3);
  const v8bf c0 = *(const v8bf*)(base + m * stride + kb);
  const v8bf c1 = *(const v8bf*)(base + m * stride + kb + 16);
  return __builtin_shufflevector(c0, c1, 0, 1, 2, 3, 4, 5, 6, 7,
                                 8, 9, 10, 11, 12, 13, 14, 15);
}

__global__ __launch_bounds__(kTHREADS)
void vanilla_lstm_kernel(const float* __restrict__ x,    // [B,T,2]
                         const float* __restrict__ We,   // [128,2]
                         const float* __restrict__ be,   // [128]
                         const float* __restrict__ Wih,  // [512,128]
                         const float* __restrict__ Whh,  // [512,128]
                         const float* __restrict__ bih,  // [512]
                         const float* __restrict__ bhh,  // [512]
                         const float* __restrict__ Wo,   // [5,128]
                         const float* __restrict__ bo,   // [5]
                         float* __restrict__ out)        // y | h | c
{
  __shared__ __align__(16) __bf16 e_sh[kROWS][kEMD + 8];   // embed, bf16
  __shared__ __align__(16) __bf16 h_sh[kROWS][kCELL + 8];  // h, bf16 (A matrix)
  __shared__ __align__(16) float  hf_sh[kROWS][kCELL];     // h, f32
  __shared__ __align__(16) float  c_sh[kROWS][kCELL];      // c, f32
  __shared__ __align__(16) float  g_sh[kROWS][kG + 4];     // gates, f32
  __shared__ float We_sh[kEMD][2];
  __shared__ float be_sh[kEMD];
  __shared__ float bsum[kG];                               // b_ih + b_hh
  __shared__ float Wo_sh[kDOUT][kCELL];
  __shared__ float bo_sh[kDOUT];

  const int tid   = threadIdx.x;
  const int lane  = tid & 31;
  const int wave  = tid >> 5;
  const int rbase = blockIdx.x * kROWS;
  const int ncol0 = wave * 32;            // this wave's gate-column base

  // ---------------- one-time init ----------------
  {
    const int r = tid >> 5, cb = (tid & 31) * 4;   // 16 rows x 128, 4/thread
    #pragma unroll
    for (int i = 0; i < 4; ++i) {
      h_sh[r][cb + i]  = (__bf16)0.0f;
      hf_sh[r][cb + i] = 0.0f;
      c_sh[r][cb + i]  = 0.0f;
    }
  }
  if (tid < kG)   bsum[tid] = bih[tid] + bhh[tid];
  if (tid < kEMD) {
    We_sh[tid][0] = We[tid * 2 + 0];
    We_sh[tid][1] = We[tid * 2 + 1];
    be_sh[tid]    = be[tid];
  }
  for (int i = tid; i < kDOUT * kCELL; i += kTHREADS)
    Wo_sh[i / kCELL][i % kCELL] = Wo[i];
  if (tid < kDOUT) bo_sh[tid] = bo[tid];

  // Pin loop-invariant weight B-fragments in VGPRs.
  // B-matrix 32x16 bf16 layout: lane L holds column N=L&15,
  // K in {kt*32+(L>>4)*8+0..7, kt*32+16+(L>>4)*8+0..7} (matches A).
  v16bf Bih[2][4], Bhh[2][4];   // [ntile][ktile]
  #pragma unroll
  for (int j = 0; j < 2; ++j) {
    const int n = ncol0 + j * 16 + (lane & 15);
    #pragma unroll
    for (int kt = 0; kt < 4; ++kt) {
      const int k0 = kt * 32 + ((lane >> 4) << 3);
      const float* wi = Wih + n * kEMD + k0;
      const float* wh = Whh + n * kCELL + k0;
      v16bf bi, bh;
      #pragma unroll
      for (int i = 0; i < 16; ++i) {
        bi[i] = (__bf16)wi[i < 8 ? i : i + 8];
        bh[i] = (__bf16)wh[i < 8 ? i : i + 8];
      }
      Bih[j][kt] = bi;
      Bhh[j][kt] = bh;
    }
  }
  __syncthreads();

  // ---------------- time loop (recurrence stays on-chip) ----------------
  for (int t = 0; t < kT; ++t) {
    // (a) embed MLP: e = relu(x @ We.T + be), written to LDS as bf16
    {
      const int r = tid >> 5, kb = (tid & 31) * 4;
      const float x0 = x[((long long)(rbase + r) * kT + t) * kDIN + 0];
      const float x1 = x[((long long)(rbase + r) * kT + t) * kDIN + 1];
      #pragma unroll
      for (int i = 0; i < 4; ++i) {
        const int k = kb + i;
        const float v = x0 * We_sh[k][0] + x1 * We_sh[k][1] + be_sh[k];
        e_sh[r][k] = (__bf16)fmaxf(v, 0.0f);
      }
    }
    __syncthreads();

    // (b) gates = e @ Wih.T + h @ Whh.T + (bih+bhh), via WMMA bf16 -> f32
    v8f acc[2];
    #pragma unroll
    for (int j = 0; j < 2; ++j) {
      const float bb = bsum[ncol0 + j * 16 + (lane & 15)];
      #pragma unroll
      for (int v = 0; v < 8; ++v) acc[j][v] = bb;
    }
    #pragma unroll
    for (int kt = 0; kt < 4; ++kt) {
      const v16bf a = load_a16(&e_sh[0][0], kEMD + 8, kt, lane);
      #pragma unroll
      for (int j = 0; j < 2; ++j)
        acc[j] = __builtin_amdgcn_wmma_f32_16x16x32_bf16(
            false, a, false, Bih[j][kt], (short)0, acc[j], false, false);
    }
    #pragma unroll
    for (int kt = 0; kt < 4; ++kt) {
      const v16bf a = load_a16(&h_sh[0][0], kCELL + 8, kt, lane);
      #pragma unroll
      for (int j = 0; j < 2; ++j)
        acc[j] = __builtin_amdgcn_wmma_f32_16x16x32_bf16(
            false, a, false, Bhh[j][kt], (short)0, acc[j], false, false);
    }
    // C/D layout: VGPR v -> row v + (lane>=16 ? 8 : 0), col = lane&15
    {
      const int mofs = (lane >> 4) << 3;
      #pragma unroll
      for (int j = 0; j < 2; ++j) {
        const int col = ncol0 + j * 16 + (lane & 15);
        #pragma unroll
        for (int v = 0; v < 8; ++v) g_sh[mofs + v][col] = acc[j][v];
      }
    }
    __syncthreads();

    // (c) LSTM pointwise: i,f,o=sigmoid, g=tanh; c' = f*c + i*g; h' = o*tanh(c')
    {
      const int r = tid >> 5, cb = (tid & 31) * 4;
      #pragma unroll
      for (int i = 0; i < 4; ++i) {
        const int cc = cb + i;
        const float gi = fast_sigmoid(g_sh[r][cc]);
        const float gf = fast_sigmoid(g_sh[r][kCELL + cc]);
        const float gg = fast_tanh(g_sh[r][2 * kCELL + cc]);
        const float go = fast_sigmoid(g_sh[r][3 * kCELL + cc]);
        const float cn = gf * c_sh[r][cc] + gi * gg;
        const float hn = go * fast_tanh(cn);
        c_sh[r][cc]  = cn;
        hf_sh[r][cc] = hn;
        h_sh[r][cc]  = (__bf16)hn;
      }
    }
    __syncthreads();

    // (d) output projection: y = h' @ Wo.T + bo  (16x5).
    // 4 threads per (row, out) dot product: 32 FMAs each, then quad-reduce
    // with __shfl_xor (wave32 cross-lane, no extra barrier).
    if (tid < kROWS * kDOUT * 4) {
      const int p   = tid >> 2;        // 0..79 : (row, out) pair
      const int sub = tid & 3;         // 0..3  : K-quarter
      const int r = p / kDOUT, o = p % kDOUT;
      const int k0 = sub * 32;
      float s = 0.0f;
      #pragma unroll 8
      for (int k = 0; k < 32; ++k) s += hf_sh[r][k0 + k] * Wo_sh[o][k0 + k];
      s += __shfl_xor(s, 1);
      s += __shfl_xor(s, 2);
      if (sub == 0)
        out[((long long)(rbase + r) * kT + t) * kDOUT + o] = s + bo_sh[o];
    }
    // final h, c
    if (t == kT - 1) {
      const long long hbase = (long long)kB * kT * kDOUT;
      const long long cbase = hbase + (long long)kB * kCELL;
      const int r = tid >> 5, cb = (tid & 31) * 4;
      #pragma unroll
      for (int i = 0; i < 4; ++i) {
        out[hbase + (long long)(rbase + r) * kCELL + cb + i] = hf_sh[r][cb + i];
        out[cbase + (long long)(rbase + r) * kCELL + cb + i] = c_sh[r][cb + i];
      }
    }
    __syncthreads();
  }
}

extern "C" void kernel_launch(void* const* d_in, const int* in_sizes, int n_in,
                              void* d_out, int out_size, void* d_ws, size_t ws_size,
                              hipStream_t stream) {
  (void)in_sizes; (void)n_in; (void)out_size; (void)d_ws; (void)ws_size;
  const float* x   = (const float*)d_in[0];
  const float* We  = (const float*)d_in[1];
  const float* be  = (const float*)d_in[2];
  const float* Wih = (const float*)d_in[3];
  const float* Whh = (const float*)d_in[4];
  const float* bih = (const float*)d_in[5];
  const float* bhh = (const float*)d_in[6];
  const float* Wo  = (const float*)d_in[7];
  const float* bo  = (const float*)d_in[8];
  float* out = (float*)d_out;

  dim3 grid(kB / kROWS);     // 256 workgroups
  dim3 block(kTHREADS);      // 512 threads = 16 wave32
  vanilla_lstm_kernel<<<grid, block, 0, stream>>>(x, We, be, Wih, Whh, bih, bhh,
                                                  Wo, bo, out);
}
